// PointNetSetAbstraction_68642167324874
// MI455X (gfx1250) — matrix operand order repaired and checked by reading, not compiled, not run on hardware
//
#include <hip/hip_runtime.h>
#include <hip/hip_bf16.h>

// ---------------- problem constants ----------------
#define BB      4
#define NN      16384
#define SS      4096        // NPOINT
#define KK      32          // NSAMPLE
#define MM      (BB * SS * KK)   // 524288 grouped points
#define R2      0.04f       // RADIUS^2
#define BIGF    1e10f
#define EPSF    1e-5f

typedef _Float16 half16 __attribute__((ext_vector_type(16)));
typedef _Float16 half8  __attribute__((ext_vector_type(8)));
typedef float    float8 __attribute__((ext_vector_type(8)));

// ---------------- weight prep + stats zero ----------------
__global__ __launch_bounds__(256) void psa_prep(const float* __restrict__ w0,
                                                const float* __restrict__ w1,
                                                const float* __restrict__ w2,
                                                _Float16* __restrict__ W0h,
                                                _Float16* __restrict__ W1h,
                                                _Float16* __restrict__ W2h,
                                                float* __restrict__ stats) {
    int tid = threadIdx.x;
    for (int i = tid; i < 64 * 32; i += 256) {           // pad 9 -> 32
        int o = i >> 5, k = i & 31;
        W0h[i] = (k < 9) ? (_Float16)w0[o * 9 + k] : (_Float16)0.0f;
    }
    for (int i = tid; i < 64 * 64; i += 256)  W1h[i] = (_Float16)w1[i];
    for (int i = tid; i < 128 * 64; i += 256) W2h[i] = (_Float16)w2[i];
    for (int i = tid; i < 512; i += 256)      stats[i] = 0.0f;  // sums/sumsq
}

// ---------------- farthest point sampling ----------------
__global__ __launch_bounds__(256) void psa_fps(const float* __restrict__ xyz,
                                               float* __restrict__ dist,
                                               int* __restrict__ cents) {
    __shared__ float sd[256];
    __shared__ int   si[256];
    __shared__ int   sFar;
    const int b = blockIdx.x, tid = threadIdx.x;
    const float* px = xyz + (size_t)b * 3 * NN;
    const float* py = px + NN;
    const float* pz = py + NN;
    float* dl = dist + (size_t)b * NN;
    for (int n = tid; n < NN; n += 256) dl[n] = BIGF;
    if (tid == 0) sFar = 0;
    __syncthreads();
    for (int it = 0; it < SS; ++it) {
        int far = sFar;
        if (tid == 0) cents[b * SS + it] = far;
        float cx = px[far], cy = py[far], cz = pz[far];
        float best = -1.0f; int bi = 0;
        for (int n = tid; n < NN; n += 256) {
            float dx = px[n] - cx, dy = py[n] - cy, dz = pz[n] - cz;
            float d = fminf(dl[n], dx * dx + dy * dy + dz * dz);
            dl[n] = d;
            if (d > best) { best = d; bi = n; }
        }
        sd[tid] = best; si[tid] = bi;
        __syncthreads();
        for (int off = 128; off > 0; off >>= 1) {
            if (tid < off && sd[tid + off] > sd[tid]) {
                sd[tid] = sd[tid + off]; si[tid] = si[tid + off];
            }
            __syncthreads();
        }
        if (tid == 0) sFar = si[0];
        __syncthreads();
    }
}

// ---------------- gather centroids, write new_xyz output ----------------
__global__ __launch_bounds__(256) void psa_gather_xyz(const float* __restrict__ xyz,
                                                      const int* __restrict__ cents,
                                                      float* __restrict__ newXYZ,
                                                      float* __restrict__ outXYZ) {
    int t = blockIdx.x * blockDim.x + threadIdx.x;
    if (t >= BB * SS) return;
    int b = t / SS, s = t % SS;
    int j = cents[t];
    #pragma unroll
    for (int c = 0; c < 3; ++c) {
        float v = xyz[((size_t)b * 3 + c) * NN + j];
        newXYZ[(size_t)t * 3 + c] = v;
        outXYZ[((size_t)b * 3 + c) * SS + s] = v;
    }
}

// ---------------- ball query: one wave32 per query ----------------
__global__ __launch_bounds__(256) void psa_ballquery(const float* __restrict__ xyz,
                                                     const float* __restrict__ newXYZ,
                                                     int* __restrict__ idxBuf) {
    const int lane = threadIdx.x & 31;
    const int q = blockIdx.x * 8 + (threadIdx.x >> 5);
    if (q >= BB * SS) return;
    const int b = q / SS;
    const float qx = newXYZ[(size_t)q * 3 + 0];
    const float qy = newXYZ[(size_t)q * 3 + 1];
    const float qz = newXYZ[(size_t)q * 3 + 2];
    const float* px = xyz + (size_t)b * 3 * NN;
    const float* py = px + NN;
    const float* pz = py + NN;
    int* out = idxBuf + (size_t)q * KK;
    int count = 0;
    for (int chunk = 0; chunk < NN && count < KK; chunk += 32) {
        int n = chunk + lane;
        float dx = px[n] - qx, dy = py[n] - qy, dz = pz[n] - qz;
        float d = dx * dx + dy * dy + dz * dz;
        bool in = !(d > R2);
        unsigned mask = (unsigned)__ballot(in);
        if (in) {
            int pos = count + __popc(mask & ((1u << lane) - 1u));
            if (pos < KK) out[pos] = n;
        }
        count += __popc(mask);
    }
    int cnt = count < KK ? count : KK;
    // ensure in-wave stores are globally visible before re-reading slot 0
    asm volatile("s_wait_storecnt 0" ::: "memory");
    int first = out[0];
    if (lane >= cnt) out[lane] = first;
}

// ---------------- build grouped features (M x 32, f16, zero-padded) ----------------
__global__ __launch_bounds__(256) void psa_feat(const float* __restrict__ xyz,
                                                const float* __restrict__ points,
                                                const float* __restrict__ newXYZ,
                                                const int* __restrict__ idxBuf,
                                                _Float16* __restrict__ feat) {
    int m = blockIdx.x * blockDim.x + threadIdx.x;
    if (m >= MM) return;
    int q = m >> 5;                 // (b*S + s)
    int b = q / SS;
    int j = idxBuf[m];
    float r0 = xyz[((size_t)b * 3 + 0) * NN + j] - newXYZ[(size_t)q * 3 + 0];
    float r1 = xyz[((size_t)b * 3 + 1) * NN + j] - newXYZ[(size_t)q * 3 + 1];
    float r2 = xyz[((size_t)b * 3 + 2) * NN + j] - newXYZ[(size_t)q * 3 + 2];
    float p[6];
    #pragma unroll
    for (int c = 0; c < 6; ++c) p[c] = points[((size_t)b * 6 + c) * NN + j];
    half8 f0 = { (_Float16)r0, (_Float16)r1, (_Float16)r2, (_Float16)p[0],
                 (_Float16)p[1], (_Float16)p[2], (_Float16)p[3], (_Float16)p[4] };
    half8 f1 = { (_Float16)p[5], (_Float16)0.f, (_Float16)0.f, (_Float16)0.f,
                 (_Float16)0.f, (_Float16)0.f, (_Float16)0.f, (_Float16)0.f };
    half8 fz = { (_Float16)0.f, (_Float16)0.f, (_Float16)0.f, (_Float16)0.f,
                 (_Float16)0.f, (_Float16)0.f, (_Float16)0.f, (_Float16)0.f };
    half8* dst = (half8*)(feat + (size_t)m * 32);
    dst[0] = f0; dst[1] = f1; dst[2] = fz; dst[3] = fz;
}

// ---------------- WMMA GEMM: Y(M,OUT) = X(M,IN) * W(OUT,IN)^T + bias; BN stats ----------
// Each wave owns one 16-channel row tile and TCOLS consecutive 16-point column
// tiles. Weights (A fragments) are loaded once; BN partial stats accumulate in
// registers across tiles so the cross-lane reduce + atomics run once per wave.
template <int IN_C, int OUT_C, int TCOLS>
__global__ __launch_bounds__(256) void psa_gemm_wmma(const _Float16* __restrict__ X,
                                                     const _Float16* __restrict__ W,
                                                     const float* __restrict__ bias,
                                                     _Float16* __restrict__ Y,
                                                     float* __restrict__ sum,
                                                     float* __restrict__ sumsq,
                                                     int Mtot) {
    const int lane = threadIdx.x & 31;
    const int gw = blockIdx.x * 8 + (threadIdx.x >> 5);
    const int rowTiles = OUT_C / 16;
    const int colGroup = gw / rowTiles;
    const int rowTile = gw % rowTiles;
    if (colGroup * 16 * TCOLS >= Mtot) return;  // wave-uniform guard (EXEC all-1 for WMMA)

    const int halfSel = lane >> 4;         // 0: lanes 0-15, 1: lanes 16-31
    const int l15 = lane & 15;
    const int row = rowTile * 16 + l15;    // output-channel row (A matrix row)
    const int chBase = rowTile * 16 + halfSel * 8;

    // A fragments: lanes 0-15 hold K {0-7,16-23}; lanes 16-31 hold K {8-15,24-31}
    half16 aFrag[IN_C / 32];
    #pragma unroll
    for (int step = 0; step < IN_C / 32; ++step) {
        const _Float16* aBase = W + (size_t)row * IN_C + step * 32 + halfSel * 8;
        half8 alo = *(const half8*)(aBase);
        half8 ahi = *(const half8*)(aBase + 16);
        aFrag[step] = __builtin_shufflevector(alo, ahi,
                          0,1,2,3,4,5,6,7,8,9,10,11,12,13,14,15);
    }
    // bias for this lane's 8 channels
    float bv[8];
    #pragma unroll
    for (int r = 0; r < 8; ++r) bv[r] = bias[chBase + r];

    float ps1[8], ps2[8];
    #pragma unroll
    for (int r = 0; r < 8; ++r) { ps1[r] = 0.0f; ps2[r] = 0.0f; }

    for (int t = 0; t < TCOLS; ++t) {
        const int mcol = (colGroup * TCOLS + t) * 16 + l15;
        const _Float16* xRow = X + (size_t)mcol * IN_C;
        // prefetch next tile's activations (global_prefetch_b8; speculative-safe)
        __builtin_prefetch(xRow + 16 * IN_C, 0, 1);

        float8 acc = {};
        #pragma unroll
        for (int step = 0; step < IN_C / 32; ++step) {
            // B fragment: lanes 0-15 hold K {0-7,16-23}; lanes 16-31 hold K {8-15,24-31} of column mcol
            const _Float16* bBase = xRow + step * 32 + halfSel * 8;
            half8 blo = *(const half8*)(bBase);
            half8 bhi = *(const half8*)(bBase + 16);
            half16 bm = __builtin_shufflevector(blo, bhi,
                          0,1,2,3,4,5,6,7,8,9,10,11,12,13,14,15);
            acc = __builtin_amdgcn_wmma_f32_16x16x32_f16(
                      false, aFrag[step], false, bm, (short)0, acc, false, false);
        }

        half8 outv;
        #pragma unroll
        for (int r = 0; r < 8; ++r) {
            float v = acc[r] + bv[r];
            outv[r] = (_Float16)v;
            ps1[r] += v;
            ps2[r] += v * v;
        }
        *(half8*)(Y + (size_t)mcol * OUT_C + chBase) = outv;
    }

    // one cross-lane reduce + atomic pair per wave (16 lanes share each channel)
    #pragma unroll
    for (int r = 0; r < 8; ++r) {
        float p1 = ps1[r], p2 = ps2[r];
        #pragma unroll
        for (int mx = 1; mx < 16; mx <<= 1) {
            p1 += __shfl_xor(p1, mx, 32);
            p2 += __shfl_xor(p2, mx, 32);
        }
        if (l15 == 0) {
            atomicAdd(&sum[chBase + r], p1);
            atomicAdd(&sumsq[chBase + r], p2);
        }
    }
}

// ---------------- finalize BN: per-channel scale/shift ----------------
__global__ void psa_finalize(const float* __restrict__ sum,
                             const float* __restrict__ sumsq,
                             const float* __restrict__ g,
                             const float* __restrict__ bt,
                             float* __restrict__ scale,
                             float* __restrict__ shift,
                             int C, float invM) {
    int c = blockIdx.x * blockDim.x + threadIdx.x;
    if (c >= C) return;
    float mu  = sum[c] * invM;
    float var = sumsq[c] * invM - mu * mu;
    float sc  = g[c] * rsqrtf(var + EPSF);
    scale[c] = sc;
    shift[c] = bt[c] - mu * sc;
}

// ---------------- in-place BN + ReLU (8 halves / thread) ----------------
template <int C>
__global__ __launch_bounds__(256) void psa_bnrelu(_Float16* __restrict__ X,
                                                  const float* __restrict__ scale,
                                                  const float* __restrict__ shift,
                                                  int Mtot) {
    int t = blockIdx.x * blockDim.x + threadIdx.x;
    int total = Mtot * (C / 8);
    if (t >= total) return;
    int c0 = (t % (C / 8)) * 8;
    half8 v = *(half8*)(X + (size_t)t * 8);
    half8 o;
    #pragma unroll
    for (int r = 0; r < 8; ++r) {
        float f = (float)v[r] * scale[c0 + r] + shift[c0 + r];
        o[r] = (_Float16)fmaxf(f, 0.0f);
    }
    *(half8*)(X + (size_t)t * 8) = o;
}

// ---------------- final BN+ReLU + max over K, write (B,128,S) ----------------
__global__ __launch_bounds__(256) void psa_maxpool(const _Float16* __restrict__ X2,
                                                   const float* __restrict__ scale,
                                                   const float* __restrict__ shift,
                                                   float* __restrict__ outPts) {
    int t = blockIdx.x * blockDim.x + threadIdx.x;
    if (t >= BB * 128 * SS) return;
    int s = t % SS;
    int o = (t / SS) & 127;
    int b = t / (SS * 128);
    const _Float16* base = X2 + ((size_t)(b * SS + s) * KK) * 128 + o;
    float sc = scale[o], sh = shift[o];
    float best = 0.0f;   // relu output is >= 0
    #pragma unroll 4
    for (int k = 0; k < KK; ++k) {
        float f = (float)base[(size_t)k * 128] * sc + sh;
        best = fmaxf(best, fmaxf(f, 0.0f));
    }
    outPts[((size_t)b * 128 + o) * SS + s] = best;
}

// ---------------- host orchestration ----------------
extern "C" void kernel_launch(void* const* d_in, const int* in_sizes, int n_in,
                              void* d_out, int out_size, void* d_ws, size_t ws_size,
                              hipStream_t stream) {
    const float* xyz    = (const float*)d_in[0];   // (4,3,16384)
    const float* points = (const float*)d_in[1];   // (4,6,16384)
    const float* w0 = (const float*)d_in[2];  const float* b0 = (const float*)d_in[3];
    const float* g0 = (const float*)d_in[4];  const float* t0 = (const float*)d_in[5];
    const float* w1 = (const float*)d_in[6];  const float* b1 = (const float*)d_in[7];
    const float* g1 = (const float*)d_in[8];  const float* t1 = (const float*)d_in[9];
    const float* w2 = (const float*)d_in[10]; const float* b2 = (const float*)d_in[11];
    const float* g2 = (const float*)d_in[12]; const float* t2 = (const float*)d_in[13];

    float* outXYZ = (float*)d_out;                 // (4,3,4096)
    float* outPts = (float*)d_out + BB * 3 * SS;   // (4,128,4096)

    // workspace carve-up (256B aligned)
    char* ws = (char*)d_ws;
    size_t off = 0;
    auto alloc = [&](size_t bytes) { char* p = ws + off; off = (off + bytes + 255) & ~(size_t)255; return p; };
    float*    dist   = (float*)   alloc((size_t)BB * NN * 4);
    int*      cents  = (int*)     alloc((size_t)BB * SS * 4);
    float*    newXYZ = (float*)   alloc((size_t)BB * SS * 3 * 4);
    int*      idxBuf = (int*)     alloc((size_t)BB * SS * KK * 4);
    _Float16* W0h    = (_Float16*)alloc(64 * 32 * 2);
    _Float16* W1h    = (_Float16*)alloc(64 * 64 * 2);
    _Float16* W2h    = (_Float16*)alloc(128 * 64 * 2);
    float*    stats  = (float*)   alloc(512 * 4);   // sum0,sq0,sum1,sq1,sum2,sq2
    float*    scsh   = (float*)   alloc(512 * 4);   // sc0,sh0,sc1,sh1,sc2,sh2
    _Float16* feat   = (_Float16*)alloc((size_t)MM * 32 * 2);
    _Float16* x0     = (_Float16*)alloc((size_t)MM * 64 * 2);
    _Float16* x1     = (_Float16*)alloc((size_t)MM * 64 * 2);
    _Float16* x2     = (_Float16*)alloc((size_t)MM * 128 * 2);
    (void)ws_size; (void)in_sizes; (void)n_in; (void)out_size;

    float* sum0 = stats;       float* sq0 = stats + 64;
    float* sum1 = stats + 128; float* sq1 = stats + 192;
    float* sum2 = stats + 256; float* sq2 = stats + 384;
    float* sc0 = scsh;         float* sh0 = scsh + 64;
    float* sc1 = scsh + 128;   float* sh1 = scsh + 192;
    float* sc2 = scsh + 256;   float* sh2 = scsh + 384;

    const float invM = 1.0f / (float)MM;
    constexpr int TC = 8;   // column tiles per wave

    psa_prep<<<1, 256, 0, stream>>>(w0, w1, w2, W0h, W1h, W2h, stats);
    psa_fps<<<BB, 256, 0, stream>>>(xyz, dist, cents);
    psa_gather_xyz<<<(BB * SS + 255) / 256, 256, 0, stream>>>(xyz, cents, newXYZ, outXYZ);
    psa_ballquery<<<(BB * SS) / 8, 256, 0, stream>>>(xyz, newXYZ, idxBuf);
    psa_feat<<<MM / 256, 256, 0, stream>>>(xyz, points, newXYZ, idxBuf, feat);

    // layer 0: 32(padded 9) -> 64
    psa_gemm_wmma<32, 64, TC><<<(MM / (16 * TC)) * (64 / 16) / 8, 256, 0, stream>>>(
        feat, W0h, b0, x0, sum0, sq0, MM);
    psa_finalize<<<1, 64, 0, stream>>>(sum0, sq0, g0, t0, sc0, sh0, 64, invM);
    psa_bnrelu<64><<<(MM * (64 / 8)) / 256, 256, 0, stream>>>(x0, sc0, sh0, MM);

    // layer 1: 64 -> 64
    psa_gemm_wmma<64, 64, TC><<<(MM / (16 * TC)) * (64 / 16) / 8, 256, 0, stream>>>(
        x0, W1h, b1, x1, sum1, sq1, MM);
    psa_finalize<<<1, 64, 0, stream>>>(sum1, sq1, g1, t1, sc1, sh1, 64, invM);
    psa_bnrelu<64><<<(MM * (64 / 8)) / 256, 256, 0, stream>>>(x1, sc1, sh1, MM);

    // layer 2: 64 -> 128
    psa_gemm_wmma<64, 128, TC><<<(MM / (16 * TC)) * (128 / 16) / 8, 256, 0, stream>>>(
        x1, W2h, b2, x2, sum2, sq2, MM);
    psa_finalize<<<1, 128, 0, stream>>>(sum2, sq2, g2, t2, sc2, sh2, 128, invM);

    // fused BN + ReLU + max over K -> output
    psa_maxpool<<<(BB * 128 * SS) / 256, 256, 0, stream>>>(x2, sc2, sh2, outPts);
}
